// QINCoQuantizer_386547057402
// MI455X (gfx1250) — matrix operand order repaired and checked by reading, not compiled, not run on hardware
//
#include <hip/hip_runtime.h>

#define N_  1024
#define D_  128
#define K_  256
#define M_  4
#define H_  256
#define NB_ 2
#define NWAVE 4

typedef __attribute__((ext_vector_type(16))) __bf16 v16bf;
typedef __attribute__((ext_vector_type(8)))  __bf16 v8bf;
typedef __attribute__((ext_vector_type(8)))  float  v8f;

__device__ __forceinline__ __bf16 f2bf(float f) {
  unsigned u = __float_as_uint(f);
  unsigned r = u + 0x7FFFu + ((u >> 16) & 1u);
  unsigned short h = (unsigned short)(r >> 16);
  return __builtin_bit_cast(__bf16, h);
}

__device__ __forceinline__ v16bf cat8(v8bf a, v8bf b) {
  return __builtin_shufflevector(a, b, 0,1,2,3,4,5,6,7,8,9,10,11,12,13,14,15);
}

__device__ __forceinline__ v8f wmma_bf(v16bf a, v16bf b, v8f c) {
  // D = A(16x32) * B(32x16) + C, bf16 in / f32 accumulate
  return __builtin_amdgcn_wmma_f32_16x16x32_bf16(false, a, false, b, (short)0, c, false, false);
}

// A fragment (16x32, 16-bit) from row-major matrix W (ld elems/row).
// Lane layout: lanes 0-15 hold M=lane, K={k0..k0+7, k0+16..k0+23};
//              lanes 16-31 hold M=lane-16, K={k0+8..k0+15, k0+24..k0+31}.
__device__ __forceinline__ v16bf afrag(const __bf16* W, int ld, int r0, int k0, int lane) {
  int r  = r0 + (lane & 15);
  int hb = (lane >> 4) & 1;
  const __bf16* p = W + (size_t)r * ld + k0 + hb * 8;
  v8bf lo = *(const v8bf*)(p);
  v8bf hi = *(const v8bf*)(p + 16);
  return cat8(lo, hi);
}

// B fragment (32x16, 16-bit): element (K=kk, col=c) = S[c][kk] (S row-major).
// Lanes 0-15: col=lane, K={k0..k0+7, k0+16..k0+23} ; lanes 16-31: col=lane-16, K={k0+8..k0+15, k0+24..k0+31}.
__device__ __forceinline__ v16bf bfrag(const __bf16* S, int ld, int c0, int k0, int lane) {
  int c  = c0 + (lane & 15);
  int kk = k0 + ((lane >> 4) & 1) * 8;
  const __bf16* p = S + (size_t)c * ld + kk;
  v8bf lo = *(const v8bf*)(p);
  v8bf hi = *(const v8bf*)(p + 16);
  return cat8(lo, hi);
}

// Runs the NB residual MLP blocks on 16 (n,k)-rows held in accC.
// accC layout: lane holds row r=lane&15; accC[t][v] = C[r][t*16 + (lane>=16?8:0) + v].
// sX: per-wave 16x136 bf16 scratch, sH: per-wave 16x264 bf16 scratch.
__device__ __forceinline__ void mlp_blocks(
    v8f accC[8], __bf16* sX, __bf16* sH,
    const __bf16* w1g, const __bf16* w2g,
    const float* b1g, const float* b2g, int lane)
{
  const int r  = lane & 15;
  const int hb = (lane >> 4) & 1;
  for (int i = 0; i < NB_; ++i) {
    // residual stream -> bf16 tile in LDS (contiguous 16B per lane per tile)
    #pragma unroll
    for (int t = 0; t < 8; ++t) {
      v8bf p;
      #pragma unroll
      for (int v = 0; v < 8; ++v) p[v] = f2bf(accC[t][v]);
      *(v8bf*)(sX + r * 136 + t * 16 + hb * 8) = p;
    }
    v16bf xf[4];
    #pragma unroll
    for (int jc = 0; jc < 4; ++jc) xf[jc] = bfrag(sX, 136, 0, jc * 32, lane);

    // GEMM1: pre_h[r,h] = sum_d W1[h,d] * X[r,d] + b1[h]; relu -> sH
    const __bf16* W1 = w1g + (size_t)i * H_ * D_;
    const float*  B1 = b1g + i * H_;
    for (int ht = 0; ht < 16; ++ht) {
      const float* bp = B1 + ht * 16 + hb * 8;
      float4 bl = *(const float4*)(bp);
      float4 bh = *(const float4*)(bp + 4);
      v8f acc;
      acc[0]=bl.x; acc[1]=bl.y; acc[2]=bl.z; acc[3]=bl.w;
      acc[4]=bh.x; acc[5]=bh.y; acc[6]=bh.z; acc[7]=bh.w;
      #pragma unroll
      for (int jc = 0; jc < 4; ++jc)
        acc = wmma_bf(afrag(W1, D_, ht * 16, jc * 32, lane), xf[jc], acc);
      v8bf hp;
      #pragma unroll
      for (int v = 0; v < 8; ++v) hp[v] = f2bf(fmaxf(acc[v], 0.0f));
      *(v8bf*)(sH + r * 264 + ht * 16 + hb * 8) = hp;
    }
    v16bf hf[8];
    #pragma unroll
    for (int hc = 0; hc < 8; ++hc) hf[hc] = bfrag(sH, 264, 0, hc * 32, lane);

    // GEMM2: C[r,d] += sum_h W2[d,h] * h[r,h] + b2[d]
    const __bf16* W2 = w2g + (size_t)i * D_ * H_;
    const float*  B2 = b2g + i * D_;
    #pragma unroll
    for (int t = 0; t < 8; ++t) {
      const float* bp = B2 + t * 16 + hb * 8;
      float4 bl = *(const float4*)(bp);
      float4 bh = *(const float4*)(bp + 4);
      v8f acc = accC[t];
      acc[0]+=bl.x; acc[1]+=bl.y; acc[2]+=bl.z; acc[3]+=bl.w;
      acc[4]+=bh.x; acc[5]+=bh.y; acc[6]+=bh.z; acc[7]+=bh.w;
      #pragma unroll
      for (int hc = 0; hc < 8; ++hc)
        acc = wmma_bf(afrag(W2, H_, t * 16, hc * 32, lane), hf[hc], acc);
      accC[t] = acc;
    }
  }
}

// ---------- Step kernels ----------

// A = x_hat @ Wx^T  (N x D), Bmat = cb @ Wc^T + cp_b (K x D); all via bf16 WMMA.
__global__ __launch_bounds__(128) void k_ab(
    const __bf16* __restrict__ cpwm,   // [D][2D] bf16
    const __bf16* __restrict__ xhatb,  // [N][D] bf16
    const __bf16* __restrict__ cbm,    // [K][D] bf16
    const float*  __restrict__ cpbm,   // [D]
    float* __restrict__ Abuf, float* __restrict__ Bbuf)
{
  int wave = threadIdx.x >> 5, lane = threadIdx.x & 31;
  int tile = blockIdx.x * 4 + wave;            // 0..639
  int c = lane & 15, hb = (lane >> 4) & 1;
  if (tile < 512) {                            // A tiles: dt 0..7, nt 0..63
    int dt = tile & 7, nt = tile >> 3;
    v8f acc = {};
    #pragma unroll
    for (int jc = 0; jc < 4; ++jc) {
      v16bf a = afrag(cpwm, 2 * D_, dt * 16, jc * 32, lane);       // Wx rows
      v16bf b = bfrag(xhatb, D_, nt * 16, jc * 32, lane);          // x_hat rows
      acc = wmma_bf(a, b, acc);
    }
    int n = nt * 16 + c, d0 = dt * 16 + hb * 8;
    float* p = Abuf + (size_t)n * D_ + d0;
    *(float4*)(p)     = make_float4(acc[0], acc[1], acc[2], acc[3]);
    *(float4*)(p + 4) = make_float4(acc[4], acc[5], acc[6], acc[7]);
  } else {                                     // B tiles: dt 0..7, kt 0..15
    int t2 = tile - 512;
    int dt = t2 & 7, kt = t2 >> 3;
    int d0 = dt * 16 + hb * 8;
    float4 bl = *(const float4*)(cpbm + d0);
    float4 bh = *(const float4*)(cpbm + d0 + 4);
    v8f acc;
    acc[0]=bl.x; acc[1]=bl.y; acc[2]=bl.z; acc[3]=bl.w;
    acc[4]=bh.x; acc[5]=bh.y; acc[6]=bh.z; acc[7]=bh.w;
    #pragma unroll
    for (int jc = 0; jc < 4; ++jc) {
      v16bf a = afrag(cpwm, 2 * D_, dt * 16, D_ + jc * 32, lane);  // Wc rows
      v16bf b = bfrag(cbm, D_, kt * 16, jc * 32, lane);            // cb rows
      acc = wmma_bf(a, b, acc);
    }
    int k = kt * 16 + c;
    float* p = Bbuf + (size_t)k * D_ + d0;
    *(float4*)(p)     = make_float4(acc[0], acc[1], acc[2], acc[3]);
    *(float4*)(p + 4) = make_float4(acc[4], acc[5], acc[6], acc[7]);
  }
}

// Fused decode + distance + argmin over a (16 n) x (NWAVE k) tile.
__global__ __launch_bounds__(128) void k_search(
    const float* __restrict__ Abuf, const float* __restrict__ Bbuf,
    const float* __restrict__ residm,
    const __bf16* __restrict__ w1m, const __bf16* __restrict__ w2m,
    const float* __restrict__ b1m, const float* __restrict__ b2m,
    unsigned long long* __restrict__ best)
{
  __shared__ __bf16 sX[NWAVE][16 * 136];
  __shared__ __bf16 sH[NWAVE][16 * 264];
  __shared__ float  sB[NWAVE][128];
  __shared__ unsigned long long sBest[16];

  const int tid = threadIdx.x;
  const int wave = tid >> 5, lane = tid & 31;
  const int n0 = blockIdx.x * 16;
  const int k0 = blockIdx.y * NWAVE;

  for (int e = tid; e < NWAVE * 128; e += 128)
    sB[e >> 7][e & 127] = Bbuf[(size_t)(k0 + (e >> 7)) * D_ + (e & 127)];
  if (tid < 16) sBest[tid] = ~0ull;
  __syncthreads();

  const int r = lane & 15, hb = (lane >> 4) & 1;
  const int n = n0 + r;

  v8f accC[8];
  #pragma unroll
  for (int t = 0; t < 8; ++t) {
    const int d0 = t * 16 + hb * 8;
    const float* ap = Abuf + (size_t)n * D_ + d0;
    float4 al = *(const float4*)(ap);
    float4 ah = *(const float4*)(ap + 4);
    float4 bl = *(const float4*)(&sB[wave][d0]);
    float4 bh = *(const float4*)(&sB[wave][d0 + 4]);
    accC[t][0]=al.x+bl.x; accC[t][1]=al.y+bl.y; accC[t][2]=al.z+bl.z; accC[t][3]=al.w+bl.w;
    accC[t][4]=ah.x+bh.x; accC[t][5]=ah.y+bh.y; accC[t][6]=ah.z+bh.z; accC[t][7]=ah.w+bh.w;
  }

  mlp_blocks(accC, sX[wave], sH[wave], w1m, w2m, b1m, b2m, lane);

  float dist = 0.0f;
  #pragma unroll
  for (int t = 0; t < 8; ++t) {
    const int d0 = t * 16 + hb * 8;
    const float* rp = residm + (size_t)n * D_ + d0;
    float4 rl = *(const float4*)(rp);
    float4 rh = *(const float4*)(rp + 4);
    float d;
    d = rl.x - accC[t][0]; dist += d * d;
    d = rl.y - accC[t][1]; dist += d * d;
    d = rl.z - accC[t][2]; dist += d * d;
    d = rl.w - accC[t][3]; dist += d * d;
    d = rh.x - accC[t][4]; dist += d * d;
    d = rh.y - accC[t][5]; dist += d * d;
    d = rh.z - accC[t][6]; dist += d * d;
    d = rh.w - accC[t][7]; dist += d * d;
  }
  dist += __shfl_xor(dist, 16, 32);
  if (lane < 16) {
    unsigned long long key =
        ((unsigned long long)__float_as_uint(dist) << 32) | (unsigned)(k0 + wave);
    atomicMin(&sBest[r], key);
  }
  __syncthreads();
  if (tid < 16) atomicMin(&best[n0 + tid], sBest[tid]);
}

// Re-decode the winning k per n, write codes/centroids, update x_hat (f32 + bf16).
__global__ __launch_bounds__(32) void k_select(
    const float* __restrict__ Abuf, const float* __restrict__ Bbuf,
    const __bf16* __restrict__ w1m, const __bf16* __restrict__ w2m,
    const float* __restrict__ b1m, const float* __restrict__ b2m,
    const unsigned long long* __restrict__ best,
    float* __restrict__ xhatf, __bf16* __restrict__ xhatb,
    float* __restrict__ codes, float* __restrict__ centm, int mIdx)
{
  __shared__ __bf16 sX[16 * 136];
  __shared__ __bf16 sH[16 * 264];
  __shared__ float  sB[16 * 128];
  __shared__ int    sK[16];

  const int lane = threadIdx.x;
  const int n0 = blockIdx.x * 16;
  if (lane < 16) {
    int k = (int)(best[n0 + lane] & 0xFFFFFFFFull);
    sK[lane] = k;
    codes[(size_t)(n0 + lane) * M_ + mIdx] = (float)k;
  }
  __syncthreads();
  for (int e = lane; e < 16 * 128; e += 32)
    sB[e] = Bbuf[(size_t)sK[e >> 7] * D_ + (e & 127)];
  __syncthreads();

  const int r = lane & 15, hb = (lane >> 4) & 1;
  const int n = n0 + r;
  v8f accC[8];
  #pragma unroll
  for (int t = 0; t < 8; ++t) {
    const int d0 = t * 16 + hb * 8;
    const float* ap = Abuf + (size_t)n * D_ + d0;
    #pragma unroll
    for (int v = 0; v < 8; ++v) accC[t][v] = ap[v] + sB[r * 128 + d0 + v];
  }

  mlp_blocks(accC, sX, sH, w1m, w2m, b1m, b2m, lane);

  #pragma unroll
  for (int t = 0; t < 8; ++t) {
    const int d0 = t * 16 + hb * 8;
    float* cp = centm + (size_t)n * D_ + d0;
    *(float4*)(cp)     = make_float4(accC[t][0], accC[t][1], accC[t][2], accC[t][3]);
    *(float4*)(cp + 4) = make_float4(accC[t][4], accC[t][5], accC[t][6], accC[t][7]);
    float* xp = xhatf + (size_t)n * D_ + d0;
    v8bf xb;
    #pragma unroll
    for (int v = 0; v < 8; ++v) {
      float nv = xp[v] + accC[t][v];
      xp[v] = nv;
      xb[v] = f2bf(nv);
    }
    *(v8bf*)(xhatb + (size_t)n * D_ + d0) = xb;
  }
}

// ---------- Small utility kernels ----------
__global__ void k_cvt(const float* __restrict__ s, __bf16* __restrict__ d, int n) {
  int i = blockIdx.x * 256 + threadIdx.x;
  if (i < n) d[i] = f2bf(s[i]);
}
__global__ void k_zero_xhat(float* xf, __bf16* xb) {
  int i = blockIdx.x * 256 + threadIdx.x;
  if (i < N_ * D_) { xf[i] = 0.0f; xb[i] = f2bf(0.0f); }
}
__global__ void k_initbest(unsigned long long* best) {
  int i = blockIdx.x * 256 + threadIdx.x;
  if (i < N_) best[i] = ~0ull;
}
__global__ void k_resid(const float* __restrict__ z, const float* __restrict__ xf,
                        float* __restrict__ out) {
  int i = blockIdx.x * 256 + threadIdx.x;
  if (i < N_ * D_) out[i] = z[i] - xf[i];
}
__global__ void k_finish(const float* __restrict__ xf, float* __restrict__ zq) {
  int i = blockIdx.x * 256 + threadIdx.x;
  if (i < N_ * D_) zq[i] = xf[i];
}

extern "C" void kernel_launch(void* const* d_in, const int* in_sizes, int n_in,
                              void* d_out, int out_size, void* d_ws, size_t ws_size,
                              hipStream_t stream) {
  (void)in_sizes; (void)n_in; (void)out_size; (void)ws_size;
  const float* z    = (const float*)d_in[0];
  const float* cb   = (const float*)d_in[1];
  const float* cpw  = (const float*)d_in[2];
  const float* cpb  = (const float*)d_in[3];
  const float* w1   = (const float*)d_in[4];
  const float* b1   = (const float*)d_in[5];
  const float* w2   = (const float*)d_in[6];
  const float* b2   = (const float*)d_in[7];

  float* out   = (float*)d_out;
  float* zq    = out;                               // N*D
  float* codes = out + (size_t)N_ * D_;             // N*M
  float* resid = codes + (size_t)N_ * M_;           // M*N*D
  float* cent  = resid + (size_t)M_ * N_ * D_;      // M*N*D

  char* ws = (char*)d_ws;
  size_t off = 0;
  auto take = [&](size_t bytes) { char* p = ws + off; off += (bytes + 255) & ~(size_t)255; return p; };
  __bf16* w1b  = (__bf16*)take((size_t)M_ * NB_ * H_ * D_ * 2);
  __bf16* w2b  = (__bf16*)take((size_t)M_ * NB_ * D_ * H_ * 2);
  __bf16* cpwb = (__bf16*)take((size_t)M_ * D_ * 2 * D_ * 2);
  __bf16* cbb  = (__bf16*)take((size_t)M_ * K_ * D_ * 2);
  float*  xhf  = (float*)take((size_t)N_ * D_ * 4);
  __bf16* xhb  = (__bf16*)take((size_t)N_ * D_ * 2);
  float*  Abuf = (float*)take((size_t)N_ * D_ * 4);
  float*  Bbuf = (float*)take((size_t)K_ * D_ * 4);
  unsigned long long* best = (unsigned long long*)take((size_t)N_ * 8);

  const int nW1 = M_ * NB_ * H_ * D_;   // 262144
  const int nW2 = M_ * NB_ * D_ * H_;   // 262144
  const int nCW = M_ * D_ * 2 * D_;     // 131072
  const int nCB = M_ * K_ * D_;         // 131072
  k_cvt<<<(nW1 + 255) / 256, 256, 0, stream>>>(w1, w1b, nW1);
  k_cvt<<<(nW2 + 255) / 256, 256, 0, stream>>>(w2, w2b, nW2);
  k_cvt<<<(nCW + 255) / 256, 256, 0, stream>>>(cpw, cpwb, nCW);
  k_cvt<<<(nCB + 255) / 256, 256, 0, stream>>>(cb, cbb, nCB);
  k_zero_xhat<<<(N_ * D_ + 255) / 256, 256, 0, stream>>>(xhf, xhb);

  for (int m = 0; m < M_; ++m) {
    const __bf16* w1m = w1b + (size_t)m * NB_ * H_ * D_;
    const __bf16* w2m = w2b + (size_t)m * NB_ * D_ * H_;
    const float*  b1m = b1 + (size_t)m * NB_ * H_;
    const float*  b2m = b2 + (size_t)m * NB_ * D_;
    float* residm = resid + (size_t)m * N_ * D_;

    k_ab<<<160, 128, 0, stream>>>(cpwb + (size_t)m * D_ * 2 * D_, xhb,
                                  cbb + (size_t)m * K_ * D_, cpb + (size_t)m * D_,
                                  Abuf, Bbuf);
    k_resid<<<(N_ * D_ + 255) / 256, 256, 0, stream>>>(z, xhf, residm);
    k_initbest<<<(N_ + 255) / 256, 256, 0, stream>>>(best);
    k_search<<<dim3(N_ / 16, K_ / NWAVE), 128, 0, stream>>>(
        Abuf, Bbuf, residm, w1m, w2m, b1m, b2m, best);
    k_select<<<N_ / 16, 32, 0, stream>>>(
        Abuf, Bbuf, w1m, w2m, b1m, b2m, best, xhf, xhb, codes,
        cent + (size_t)m * N_ * D_, m);
  }
  k_finish<<<(N_ * D_ + 255) / 256, 256, 0, stream>>>(xhf, zq);
}
